// RAT_8675833938323
// MI455X (gfx1250) — matrix operand order repaired and checked by reading, not compiled, not run on hardware
//
#include <hip/hip_runtime.h>

// ---------------------------------------------------------------------------
// Types for CDNA5 WMMA (wave32, 16x16x32 bf16 -> f32 accumulate)
// ---------------------------------------------------------------------------
typedef __attribute__((ext_vector_type(16))) __bf16 v16bf;
typedef __attribute__((ext_vector_type(2)))  __bf16 v2bf;
typedef __attribute__((ext_vector_type(8)))  float  v8f;

union FragBF {
    v16bf v;
    uint4 q[2];
};

__device__ __forceinline__ unsigned short f2bf(float f) {
    union { __bf16 h; unsigned short u; } c;
    c.h = (__bf16)f;                    // native f32->bf16 convert
    return c.u;
}

// Two fptruncs built into a v2bf16 -> single v_cvt_pk_bf16_f32, no repacking.
__device__ __forceinline__ unsigned pack2bf(float a, float b) {
    v2bf h;
    h[0] = (__bf16)a;
    h[1] = (__bf16)b;
    union { v2bf h; unsigned u; } c;
    c.h = h;
    return c.u;
}

__device__ __forceinline__ void store4bf(unsigned short* dst, float4 v) {
    uint2 t;
    t.x = pack2bf(v.x, v.y);
    t.y = pack2bf(v.z, v.w);
    *(uint2*)dst = t;                   // one ds_store_b64
}

// ---------------------------------------------------------------------------
// Global -> register staging (one 128x32 A tile + one B tile per block)
// ---------------------------------------------------------------------------
template<bool TRANS_B>
__device__ __forceinline__ void stage_load(
    const float* __restrict__ A, const float* __restrict__ Bm,
    int lda, int ldb, int rowBase, int colBase, int k0, int tid,
    float4 (&ra)[4], float4 (&rb)[4])
{
#pragma unroll
    for (int i = 0; i < 4; ++i) {
        int f  = tid + i * 256;          // 1024 float4 chunks
        int r  = f >> 3;                 // 8 float4 per row of 32 k
        int kc = (f & 7) << 2;
        ra[i] = *(const float4*)(A + (long long)(rowBase + r) * lda + (k0 + kc));
    }
    if (TRANS_B) {
#pragma unroll
        for (int i = 0; i < 4; ++i) {
            int f  = tid + i * 256;
            int n  = f >> 3;
            int kc = (f & 7) << 2;
            rb[i] = *(const float4*)(Bm + (long long)(colBase + n) * ldb + (k0 + kc));
        }
    } else {
#pragma unroll
        for (int i = 0; i < 4; ++i) {
            int f  = tid + i * 256;
            int kk = f >> 5;             // 32 k rows
            int nc = (f & 31) << 2;      // 128 cols / 4
            rb[i] = *(const float4*)(Bm + (long long)(k0 + kk) * ldb + (colBase + nc));
        }
    }
}

// Register -> LDS (with f32->bf16 convert). As[row][k], Bs[col][k].
template<bool TRANS_B>
__device__ __forceinline__ void stage_store(
    unsigned short* __restrict__ As, unsigned short* __restrict__ Bs,
    int tid, const float4 (&ra)[4], const float4 (&rb)[4])
{
#pragma unroll
    for (int i = 0; i < 4; ++i) {
        int f  = tid + i * 256;
        int r  = f >> 3;
        int kc = (f & 7) << 2;
        store4bf(&As[r * 32 + kc], ra[i]);
    }
    if (TRANS_B) {
#pragma unroll
        for (int i = 0; i < 4; ++i) {
            int f  = tid + i * 256;
            int n  = f >> 3;
            int kc = (f & 7) << 2;
            store4bf(&Bs[n * 32 + kc], rb[i]);
        }
    } else {
#pragma unroll
        for (int i = 0; i < 4; ++i) {
            int f  = tid + i * 256;
            int kk = f >> 5;
            int nc = (f & 31) << 2;
            Bs[(nc + 0) * 32 + kk] = f2bf(rb[i].x);
            Bs[(nc + 1) * 32 + kk] = f2bf(rb[i].y);
            Bs[(nc + 2) * 32 + kk] = f2bf(rb[i].z);
            Bs[(nc + 3) * 32 + kk] = f2bf(rb[i].w);
        }
    }
}

// ---------------------------------------------------------------------------
// Generic bf16 WMMA GEMM:  C = alpha * A x op(B)  [+ Res] [ReLU]
//   A:  [M,K]  f32 row-major, leading dim lda
//   B:  TRANS_B==0 : [K,N] row-major (ldb)   (NN)
//       TRANS_B==1 : [N,K] row-major (ldb), used transposed (NT)
//   C:  [M,N] f32 row-major, ldc.  Res indexed like C.
//   Batch: z = blockIdx.z ; z1 = z/hdiv, z2 = z%hdiv ; off = z1*s?1 + z2*s?2
// Block tile 128x128, BK=32, 256 threads (8 waves as 4(m) x 2(n)),
// wave tile 32x64 = 2x4 fragments of 16x16.
// Double-buffered LDS, one barrier per K-step; next-tile global loads are
// issued right after the barrier so they overlap the 8-WMMA burst.
// ---------------------------------------------------------------------------
template<bool TRANS_B, bool RELU, bool RESID>
__global__ __launch_bounds__(256) void gemm_bf16(
    const float* __restrict__ A, const float* __restrict__ Bm,
    float* __restrict__ C, const float* __restrict__ Res,
    int M, int Ncol, int Kd, int lda, int ldb, int ldc,
    long long sA1, long long sA2, long long sB1, long long sB2,
    long long sC1, long long sC2, long long sR1, long long sR2,
    int hdiv, float alpha)
{
    __shared__ __align__(16) unsigned short As[2][128 * 32];  // [buf][row][k]
    __shared__ __align__(16) unsigned short Bs[2][128 * 32];  // [buf][col][k]

    const int z  = blockIdx.z;
    const int z1 = z / hdiv;
    const int z2 = z - z1 * hdiv;
    A  += z1 * sA1 + z2 * sA2;
    Bm += z1 * sB1 + z2 * sB2;
    C  += z1 * sC1 + z2 * sC2;
    if (RESID) Res += z1 * sR1 + z2 * sR2;

    const int tid  = threadIdx.x;
    const int wave = tid >> 5;
    const int lane = tid & 31;
    const int wm   = wave & 3;      // 0..3  -> row block of 32
    const int wn   = wave >> 2;     // 0..1  -> col block of 64
    const int t    = lane & 15;
    const int hi   = lane >> 4;

    const int rowBase = blockIdx.y * 128;
    const int colBase = blockIdx.x * 128;

    v8f acc[2][4] = {};

    const int ktiles = Kd >> 5;     // Kd % 32 == 0 for all launches
    float4 ra[4], rb[4];

    // Prologue: stage tile 0 into buffer 0.
    stage_load<TRANS_B>(A, Bm, lda, ldb, rowBase, colBase, 0, tid, ra, rb);
    stage_store<TRANS_B>(As[0], Bs[0], tid, ra, rb);

    for (int kt = 0; kt < ktiles; ++kt) {
        __syncthreads();            // tile kt visible; prior reads of other buf done

        const bool more = (kt + 1) < ktiles;
        if (more)                   // issue next-tile global loads early
            stage_load<TRANS_B>(A, Bm, lda, ldb, rowBase, colBase,
                                (kt + 1) << 5, tid, ra, rb);

        const unsigned short* Ab = As[kt & 1];
        const unsigned short* Bb = Bs[kt & 1];

        // ---- Fragments per ISA lane layout ----
        FragBF a[2], b[4];
#pragma unroll
        for (int i = 0; i < 2; ++i) {
            int r = wm * 32 + i * 16 + t;
            const uint4* p = (const uint4*)&Ab[r * 32];
            a[i].q[0] = p[hi];       // K = hi*8 .. hi*8+7
            a[i].q[1] = p[2 + hi];   // K = 16+hi*8 .. 16+hi*8+7
        }
#pragma unroll
        for (int j = 0; j < 4; ++j) {
            int n = wn * 64 + j * 16 + t;
            const uint4* p = (const uint4*)&Bb[n * 32];
            b[j].q[0] = p[hi];           // K = hi*8 .. hi*8+7
            b[j].q[1] = p[2 + hi];       // K = 16+hi*8 .. 16+hi*8+7
        }
#pragma unroll
        for (int i = 0; i < 2; ++i)
#pragma unroll
            for (int j = 0; j < 4; ++j)
                acc[i][j] = __builtin_amdgcn_wmma_f32_16x16x32_bf16(
                    false, a[i].v, false, b[j].v,
                    (short)0, acc[i][j], false, false);

        if (more)                   // fill the other buffer for next iteration
            stage_store<TRANS_B>(As[(kt + 1) & 1], Bs[(kt + 1) & 1], tid, ra, rb);
    }

    // ---- Epilogue: C/D layout: row = hi*8 + r, col = t ----
#pragma unroll
    for (int i = 0; i < 2; ++i) {
#pragma unroll
        for (int j = 0; j < 4; ++j) {
            int col = colBase + wn * 64 + j * 16 + t;
#pragma unroll
            for (int r = 0; r < 8; ++r) {
                int row = rowBase + wm * 32 + i * 16 + hi * 8 + r;
                float v = acc[i][j][r] * alpha;
                if (RESID) v += Res[(long long)row * ldc + col];
                if (RELU)  v = fmaxf(v, 0.0f);
                C[(long long)row * ldc + col] = v;
            }
        }
    }
}

// ---------------------------------------------------------------------------
// Row softmax, rows of exactly 1024 f32, in place. 256 threads x float4.
// ---------------------------------------------------------------------------
__global__ __launch_bounds__(256) void softmax1024(float* __restrict__ d)
{
    const int tid = threadIdx.x;
    float* row = d + (long long)blockIdx.x * 1024;
    float4 v = ((float4*)row)[tid];

    __shared__ float red[256];

    float m = fmaxf(fmaxf(v.x, v.y), fmaxf(v.z, v.w));
    red[tid] = m; __syncthreads();
    for (int s = 128; s > 0; s >>= 1) {
        if (tid < s) red[tid] = fmaxf(red[tid], red[tid + s]);
        __syncthreads();
    }
    m = red[0]; __syncthreads();

    float4 e;
    e.x = __expf(v.x - m); e.y = __expf(v.y - m);
    e.z = __expf(v.z - m); e.w = __expf(v.w - m);

    red[tid] = e.x + e.y + e.z + e.w; __syncthreads();
    for (int s = 128; s > 0; s >>= 1) {
        if (tid < s) red[tid] += red[tid + s];
        __syncthreads();
    }
    float inv = 1.0f / red[0];

    e.x *= inv; e.y *= inv; e.z *= inv; e.w *= inv;
    ((float4*)row)[tid] = e;
}

// ---------------------------------------------------------------------------
// out = LayerNorm(x [+ add]) over rows of 1024, with gamma/beta.
// ---------------------------------------------------------------------------
__global__ __launch_bounds__(256) void add_layernorm1024(
    const float* __restrict__ x, const float* __restrict__ add,
    const float* __restrict__ g, const float* __restrict__ b,
    float* __restrict__ out, float eps)
{
    const int tid = threadIdx.x;
    const long long base = (long long)blockIdx.x * 1024;

    float4 v = ((const float4*)(x + base))[tid];
    if (add) {
        float4 a = ((const float4*)(add + base))[tid];
        v.x += a.x; v.y += a.y; v.z += a.z; v.w += a.w;
    }

    __shared__ float rs[256];
    __shared__ float rq[256];

    rs[tid] = v.x + v.y + v.z + v.w;
    rq[tid] = v.x * v.x + v.y * v.y + v.z * v.z + v.w * v.w;
    __syncthreads();
    for (int s = 128; s > 0; s >>= 1) {
        if (tid < s) { rs[tid] += rs[tid + s]; rq[tid] += rq[tid + s]; }
        __syncthreads();
    }
    const float inv_n = 1.0f / 1024.0f;
    float mu  = rs[0] * inv_n;
    float var = rq[0] * inv_n - mu * mu;
    float rstd = rsqrtf(var + eps);

    float4 gg = ((const float4*)g)[tid];
    float4 bb = ((const float4*)b)[tid];
    float4 o;
    o.x = (v.x - mu) * rstd * gg.x + bb.x;
    o.y = (v.y - mu) * rstd * gg.y + bb.y;
    o.z = (v.z - mu) * rstd * gg.z + bb.z;
    o.w = (v.w - mu) * rstd * gg.w + bb.w;
    ((float4*)(out + base))[tid] = o;
}

// ---------------------------------------------------------------------------
// Host-side orchestration
// ---------------------------------------------------------------------------
extern "C" void kernel_launch(void* const* d_in, const int* in_sizes, int n_in,
                              void* d_out, int out_size, void* d_ws, size_t ws_size,
                              hipStream_t stream)
{
    (void)in_sizes; (void)n_in; (void)out_size; (void)ws_size;

    const float* Fs   = (const float*)d_in[0];
    const float* Fq   = (const float*)d_in[1];
    const float* Fms  = (const float*)d_in[2];
    // d_in[3] = out_dim scalar (N), unused
    const float* W1   = (const float*)d_in[4];
    const float* W2   = (const float*)d_in[5];
    const float* W3   = (const float*)d_in[6];
    const float* W4   = (const float*)d_in[7];
    const float* W5   = (const float*)d_in[8];
    const float* W6   = (const float*)d_in[9];
    const float* Wq   = (const float*)d_in[10];
    const float* Wk   = (const float*)d_in[11];
    const float* Wv   = (const float*)d_in[12];
    const float* ln_g = (const float*)d_in[13];
    const float* ln_b = (const float*)d_in[14];
    const float* fw1  = (const float*)d_in[15];
    const float* fw2  = (const float*)d_in[16];
    const float* fln_g= (const float*)d_in[17];
    const float* fln_b= (const float*)d_in[18];

    constexpr int Bb = 16, N = 1024, FEAT = 512, HID = 1024, HEADS = 2, G = 512;
    const float inv_sqrt_feat = 0.04419417382415922f;  // 1/sqrt(512)
    const float inv_sqrt_g    = 0.04419417382415922f;  // g = 512

    const size_t MB = 1ull << 20;
    char* w = (char*)d_ws;
    float* Pa   = (float*)(w + 0);          //  32 MB : [B,N,FEAT]
    float* Pb   = (float*)(w + 32  * MB);   //  32 MB
    float* ass  = (float*)(w + 64  * MB);   //  64 MB : [B,N,N]
    float* asq  = (float*)(w + 128 * MB);   //  64 MB
    float* aqq  = (float*)(w + 192 * MB);   //  64 MB
    float* Qb   = (float*)(w + 256 * MB);   //  64 MB
    float* Kb   = (float*)(w + 320 * MB);   //  64 MB
    float* Vb   = (float*)(w + 384 * MB);   //  64 MB
    float* S    = ass;                      // 128 MB overlay (ass+asq) for scores
    float* o    = Qb;                       // reuse after scores
    float* x    = Kb;                       // reuse after attention
    float* h    = Vb;                       // reuse after o
    float* ypre = ass;                      // reuse after o

    const dim3 blk(256, 1, 1);
    const long long NF = (long long)N * FEAT;
    const long long NN = (long long)N * N;

    // ---- Feature projections: (B*N, FEAT) @ (FEAT, FEAT) ----
    const dim3 gProj(FEAT / 128, (Bb * N) / 128, 1);
#define PROJ(AA, WW, OO) \
    gemm_bf16<false,false,false><<<gProj, blk, 0, stream>>>( \
        AA, WW, OO, nullptr, Bb*N, FEAT, FEAT, FEAT, FEAT, FEAT, \
        0,0,0,0,0,0,0,0, 1, 1.0f)

    // ---- Affinity pattern: softmax( (aWa)(bWb)^T / sqrt(FEAT) ) ----
    const dim3 gAff(N / 128, N / 128, Bb);
#define AFFIN(OO) \
    gemm_bf16<true,false,false><<<gAff, blk, 0, stream>>>( \
        Pa, Pb, OO, nullptr, N, N, FEAT, FEAT, FEAT, N, \
        NF, 0, NF, 0, NN, 0, 0, 0, 1, inv_sqrt_feat); \
    softmax1024<<<Bb * N, blk, 0, stream>>>(OO)

    PROJ(Fms, W1, Pa); PROJ(Fms, W2, Pb); AFFIN(ass);
    PROJ(Fs,  W3, Pa); PROJ(Fq,  W4, Pb); AFFIN(asq);
    PROJ(Fq,  W5, Pa); PROJ(Fq,  W6, Pb); AFFIN(aqq);

    // ---- Q, K, V : (B*N, N) @ (N, N) shared weights ----
    const dim3 gQKV(N / 128, (Bb * N) / 128, 1);
#define QKV(AA, WW, OO) \
    gemm_bf16<false,false,false><<<gQKV, blk, 0, stream>>>( \
        AA, WW, OO, nullptr, Bb*N, N, N, N, N, N, \
        0,0,0,0,0,0,0,0, 1, 1.0f)
    QKV(asq, Wq, Qb);
    QKV(ass, Wk, Kb);
    QKV(asq, Wv, Vb);

    // ---- Attention scores: per (b,h): Qs @ Ks^T / sqrt(g) ----
    gemm_bf16<true,false,false><<<dim3(N / 128, N / 128, Bb * HEADS), blk, 0, stream>>>(
        Qb, Kb, S, nullptr, N, N, G, N, N, N,
        NN, G, NN, G, (long long)HEADS * NN, NN, 0, 0, HEADS, inv_sqrt_g);
    softmax1024<<<Bb * HEADS * N, blk, 0, stream>>>(S);

    // ---- o = attn @ Vs  (writes head slices of [B,N,N]) ----
    gemm_bf16<false,false,false><<<dim3(G / 128, N / 128, Bb * HEADS), blk, 0, stream>>>(
        S, Vb, o, nullptr, N, G, N, N, N, N,
        (long long)HEADS * NN, NN, NN, G, NN, G, 0, 0, HEADS, 1.0f);

    // ---- x = LN(o + aqq) ----
    add_layernorm1024<<<Bb * N, blk, 0, stream>>>(o, aqq, ln_g, ln_b, x, 1e-5f);

    // ---- FFN: h = relu(x @ w1) ; ypre = h @ w2 + x ----
    gemm_bf16<false,true,false><<<dim3(HID / 128, (Bb * N) / 128, 1), blk, 0, stream>>>(
        x, fw1, h, nullptr, Bb*N, HID, N, N, HID, HID,
        0,0,0,0,0,0,0,0, 1, 1.0f);
    gemm_bf16<false,false,true><<<dim3(N / 128, (Bb * N) / 128, 1), blk, 0, stream>>>(
        h, fw2, ypre, x, Bb*N, N, HID, HID, N, N,
        0,0,0,0,0,0,0,0, 1, 1.0f);

    // ---- y = LN(ypre) -> d_out ----
    add_layernorm1024<<<Bb * N, blk, 0, stream>>>(ypre, nullptr, fln_g, fln_b,
                                                  (float*)d_out, 1e-6f);
#undef PROJ
#undef AFFIN
#undef QKV
}
